// EGNNLayer_41961830482382
// MI455X (gfx1250) — matrix operand order repaired and checked by reading, not compiled, not run on hardware
//
#include <hip/hip_runtime.h>
#include <hip/hip_bf16.h>

typedef __attribute__((ext_vector_type(16))) _Float16 v16h;
typedef __attribute__((ext_vector_type(8)))  float    v8f;

#define BB 4
#define NN 512
#define HFEAT 64
#define MMSG 32
#define BN (BB * NN)

// ---------------------------------------------------------------------------
// D = A x B + C, 16x16x32 f16 -> f32 accum (CDNA5 WMMA)
__device__ __forceinline__ v8f wmma16(v16h a, v16h b, v8f c) {
  return __builtin_amdgcn_wmma_f32_16x16x32_f16(false, a, false, b, (short)0, c,
                                                false, false);
}

// Build an A-matrix tile (16x32 f16) holding A[m][k] = W[k0+k][n0+m]
// (i.e. transposed weight W^T), guarded for padded columns.
// A layout: lane m=lane&15, half h=lane>>4; VGPR p packs
// k_local = 16*h + 2*p + {0,1}.
__device__ __forceinline__ v16h load_AT(const float* W, int ldn, int ncols,
                                        int k0, int n0, int lane) {
  const int m = lane & 15, h2 = lane >> 4;
  const bool ok = (n0 + m) < ncols;
  v16h a;
#pragma unroll
  for (int p = 0; p < 8; ++p) {
    const int kl = 16 * h2 + 2 * p;
    float v0 = ok ? W[(size_t)(k0 + kl) * ldn + n0 + m] : 0.f;
    float v1 = ok ? W[(size_t)(k0 + kl + 1) * ldn + n0 + m] : 0.f;
    a[2 * p] = (_Float16)v0;
    a[2 * p + 1] = (_Float16)v1;
  }
  return a;
}

// C/D layout: lane col n=lane&15, VGPR v holds row M = v + 8*(lane>>4).
// Broadcast a bias vector over rows (zero-padded).
__device__ __forceinline__ v8f bias_c(const float* bv, int nb, int n0, int lane) {
  const int h2 = lane >> 4;
  v8f c;
#pragma unroll
  for (int v = 0; v < 8; ++v) {
    int o = n0 + 8 * h2 + v;
    c[v] = (o < nb) ? bv[o] : 0.f;
  }
  return c;
}

// Pack 16 contiguous f32 (this lane's B-layout run: K = 16*kh + 0..15) to f16.
__device__ __forceinline__ v16h pack16(const float* p) {
  v16h r;
#pragma unroll
  for (int q = 0; q < 16; ++q) r[q] = (_Float16)p[q];
  return r;
}

// Packed-f16 relu (lowers to v_pk_max_num_f16).
__device__ __forceinline__ v16h h_relu(v16h v) {
#pragma unroll
  for (int q = 0; q < 16; ++q) v[q] = v[q] > (_Float16)0.f ? v[q] : (_Float16)0.f;
  return v;
}

// ---------------------------------------------------------------------------
// Phase 0: per-node projections of the message MLP first layer, stored f16
// (they feed the f16 WMMA B operand anyway).
//   A_pre[row,k] = h[row,:] @ msg_w1[0:64 ,k] + msg_b1[k]   (h_i half, bias folded)
//   B_pre[row,k] = h[row,:] @ msg_w1[64:128,k]              (h_j half)
__global__ void egnn_pre(const float* __restrict__ h,
                         const float* __restrict__ msg_w1,
                         const float* __restrict__ msg_b1,
                         _Float16* __restrict__ A_pre,
                         _Float16* __restrict__ B_pre) {
  const int row = blockIdx.x;       // 0 .. BN-1
  const int k = threadIdx.x;        // 0 .. 63
  const float* hr = h + (size_t)row * HFEAT;
  float a = msg_b1[k];
  float bacc = 0.f;
#pragma unroll 8
  for (int c = 0; c < HFEAT; ++c) {
    float hv = hr[c];
    a += hv * msg_w1[c * HFEAT + k];
    bacc += hv * msg_w1[(HFEAT + c) * HFEAT + k];
  }
  A_pre[(size_t)row * HFEAT + k] = (_Float16)a;
  B_pre[(size_t)row * HFEAT + k] = (_Float16)bacc;
}

// ---------------------------------------------------------------------------
// Main kernel: one block = one (batch, 16-row i-block). 8 waves stride over j.
__global__ __launch_bounds__(256) void egnn_main(
    const float* __restrict__ x, const float* __restrict__ h,
    const unsigned char* __restrict__ mask,
    const float* __restrict__ msg_w1, const float* __restrict__ msg_w2,
    const float* __restrict__ msg_b2, const float* __restrict__ pos_w1,
    const float* __restrict__ pos_b1, const float* __restrict__ pos_w2,
    const float* __restrict__ pos_b2, const float* __restrict__ feat_w1,
    const float* __restrict__ feat_b1, const float* __restrict__ feat_w2,
    const float* __restrict__ feat_b2, const _Float16* __restrict__ A_pre,
    const _Float16* __restrict__ B_pre, float* __restrict__ x_out,
    float* __restrict__ h_out) {
  const int tid = threadIdx.x;
  const int wave = tid >> 5, lane = tid & 31;
  const int n = lane & 15;     // column of D = edge row index i0+n
  const int hh = lane >> 4;    // lane half: selects K run / D row half
  const int blk = blockIdx.x;  // 0..127
  const int b = blk >> 5;      // 32 i-blocks per batch
  const int i0 = (blk & 31) << 4;
  const int bi0 = b * NN + i0;
  const int i = i0 + n;

  __shared__ __align__(16) float lds_tr[8][16 * 64];  // per-wave transpose pads
  __shared__ __align__(16) float lds_agg[16][32];
  __shared__ __align__(16) float lds_hid[16][64];
  __shared__ float lds_mx[16][3];
  __shared__ float lds_deg[16];

  for (int t = tid; t < 16 * 32; t += 256) ((float*)lds_agg)[t] = 0.f;
  if (tid < 48) ((float*)lds_mx)[tid] = 0.f;
  if (tid < 16) lds_deg[tid] = 0.f;
  __syncthreads();

  // Loop-invariant per-lane data (packed f16, already in B-tile layout) -----
  const _Float16* aprow = A_pre + (size_t)(bi0 + n) * HFEAT;
  v16h ap0 = *(const v16h*)(aprow + 16 * hh);
  v16h ap1 = *(const v16h*)(aprow + 32 + 16 * hh);
  const float* w1d2 = msg_w1 + 128 * HFEAT;  // d2 row of msg_w1
  v16h wd0, wd1;
#pragma unroll
  for (int q = 0; q < 16; ++q) {
    wd0[q] = (_Float16)w1d2[16 * hh + q];
    wd1[q] = (_Float16)w1d2[32 + 16 * hh + q];
  }
  const float xi0 = x[(size_t)(bi0 + n) * 3 + 0];
  const float xi1 = x[(size_t)(bi0 + n) * 3 + 1];
  const float xi2 = x[(size_t)(bi0 + n) * 3 + 2];
  const bool maski = mask[bi0 + n] != 0;

  // Weight A-tiles (transposed weights) and bias C-tiles, kept in VGPRs.
  v16h w2t00 = load_AT(msg_w2, 32, 32, 0, 0, lane);
  v16h w2t01 = load_AT(msg_w2, 32, 32, 32, 0, lane);
  v16h w2t10 = load_AT(msg_w2, 32, 32, 0, 16, lane);
  v16h w2t11 = load_AT(msg_w2, 32, 32, 32, 16, lane);
  v16h pw1t[4];
#pragma unroll
  for (int mt = 0; mt < 4; ++mt) pw1t[mt] = load_AT(pos_w1, 64, 64, 0, 16 * mt, lane);
  v16h pw2t0 = load_AT(pos_w2, 3, 3, 0, 0, lane);   // padded N: cols 3..15 = 0
  v16h pw2t1 = load_AT(pos_w2, 3, 3, 32, 0, lane);
  const v8f b2c0 = bias_c(msg_b2, 32, 0, lane);
  const v8f b2c1 = bias_c(msg_b2, 32, 16, lane);
  v8f pb1c[4];
#pragma unroll
  for (int mt = 0; mt < 4; ++mt) pb1c[mt] = bias_c(pos_b1, 64, 16 * mt, lane);
  const v8f pb2c = bias_c(pos_b2, 3, 0, lane);

  v8f agg0 = {}, agg1 = {};
  float mx0 = 0.f, mx1 = 0.f, mx2 = 0.f, degf = 0.f;
  float* sc = lds_tr[wave];

  // j loop: 16 edges (i-block x one j) per wave per step -------------------
  for (int j = wave; j < NN; j += 8) {
    const _Float16* bprow = B_pre + (size_t)(b * NN + j) * HFEAT;
    {
      int jn = j + 8 < NN ? j + 8 : j;
      __builtin_prefetch(B_pre + (size_t)(b * NN + jn) * HFEAT, 0, 0);
    }
    v16h bp0v = *(const v16h*)(bprow + 16 * hh);
    v16h bp1v = *(const v16h*)(bprow + 32 + 16 * hh);

    const float xj0 = x[(size_t)(b * NN + j) * 3 + 0];
    const float xj1 = x[(size_t)(b * NN + j) * 3 + 1];
    const float xj2 = x[(size_t)(b * NN + j) * 3 + 2];
    const bool maskj = mask[b * NN + j] != 0;
    const float valid = (maski && maskj && (i != j)) ? 1.f : 0.f;
    const float r0 = (xi0 - xj0) * valid;
    const float r1 = (xi1 - xj1) * valid;
    const float r2 = (xi2 - xj2) * valid;
    const float d2 = r0 * r0 + r1 * r1 + r2 * r2;
    const _Float16 d2h = (_Float16)d2;
    v16h d2v;
#pragma unroll
    for (int q = 0; q < 16; ++q) d2v[q] = d2h;

    // hidden1 (K=64) in packed f16, directly in B-tile layout
    v16h bh0 = h_relu(ap0 + bp0v + d2v * wd0);
    v16h bh1 = h_relu(ap1 + bp1v + d2v * wd1);

    // m^T = msg_w2^T @ hidden1^T  (32 out x 16 edges)
    v8f m0 = wmma16(w2t01, bh1, wmma16(w2t00, bh0, b2c0));
    v8f m1 = wmma16(w2t11, bh1, wmma16(w2t10, bh0, b2c1));
#pragma unroll
    for (int v = 0; v < 8; ++v) {
      m0[v] = fmaxf(m0[v], 0.f) * valid;  // relu + edge mask (per-edge = per-col)
      m1[v] = fmaxf(m1[v], 0.f) * valid;
    }
    agg0 += m0;
    agg1 += m1;

    // transpose m (D-layout -> B-layout) via per-wave LDS: store [edge][k]
    *(float4*)&sc[n * 32 + 8 * hh] = make_float4(m0[0], m0[1], m0[2], m0[3]);
    *(float4*)&sc[n * 32 + 8 * hh + 4] = make_float4(m0[4], m0[5], m0[6], m0[7]);
    *(float4*)&sc[n * 32 + 16 + 8 * hh] = make_float4(m1[0], m1[1], m1[2], m1[3]);
    *(float4*)&sc[n * 32 + 16 + 8 * hh + 4] = make_float4(m1[4], m1[5], m1[6], m1[7]);
    v16h bm = pack16(&sc[n * 32 + 16 * hh]);

    // ph^T = pos_w1^T @ m^T  (64 out x 16 edges), relu
    v8f ph[4];
#pragma unroll
    for (int mt = 0; mt < 4; ++mt) {
      ph[mt] = wmma16(pw1t[mt], bm, pb1c[mt]);
#pragma unroll
      for (int v = 0; v < 8; ++v) ph[mt][v] = fmaxf(ph[mt][v], 0.f);
    }
    // transpose ph via LDS: [edge][k0..63]
#pragma unroll
    for (int mt = 0; mt < 4; ++mt) {
      *(float4*)&sc[n * 64 + 16 * mt + 8 * hh] =
          make_float4(ph[mt][0], ph[mt][1], ph[mt][2], ph[mt][3]);
      *(float4*)&sc[n * 64 + 16 * mt + 8 * hh + 4] =
          make_float4(ph[mt][4], ph[mt][5], ph[mt][6], ph[mt][7]);
    }
    v16h bq0 = pack16(&sc[n * 64 + 16 * hh]);
    v16h bq1 = pack16(&sc[n * 64 + 32 + 16 * hh]);

    // s^T = pos_w2^T(pad16) @ ph^T : lanes 0-15 rows 0..2 hold s[edge][c]
    v8f s = wmma16(pw2t1, bq1, wmma16(pw2t0, bq0, pb2c));
    mx0 += s[0] * r0;  // only meaningful on lane half 0 (rows 0..2)
    mx1 += s[1] * r1;
    mx2 += s[2] * r2;
    degf += valid;
  }

  // cross-wave reduction ----------------------------------------------------
  if (lane < 16) {
    atomicAdd(&lds_mx[n][0], mx0);
    atomicAdd(&lds_mx[n][1], mx1);
    atomicAdd(&lds_mx[n][2], mx2);
    atomicAdd(&lds_deg[n], degf);
  }
#pragma unroll
  for (int v = 0; v < 8; ++v) {
    atomicAdd(&lds_agg[n][8 * hh + v], agg0[v]);
    atomicAdd(&lds_agg[n][16 + 8 * hh + v], agg1[v]);
  }
  __syncthreads();

  // x_out -------------------------------------------------------------------
  if (tid < 48) {
    int il = tid / 3, c = tid % 3;
    float dg = lds_deg[il];
    dg = dg > 0.f ? dg : 1.f;
    x_out[(size_t)(bi0 + il) * 3 + c] =
        x[(size_t)(bi0 + il) * 3 + c] + ((float*)lds_mx)[il * 3 + c] / dg;
  }

  // feature MLP: h_out^T = feat_w2^T @ relu(feat_w1^T @ [h|agg]^T + b1) + b2
  if (wave < 4) {
    const float* hrow = h + (size_t)(bi0 + n) * HFEAT;
    v16h bc0 = pack16(&hrow[16 * hh]);
    v16h bc1 = pack16(&hrow[32 + 16 * hh]);
    v16h bc2 = pack16(&lds_agg[n][16 * hh]);
    v16h fa0 = load_AT(feat_w1, 64, 64, 0, 16 * wave, lane);
    v16h fa1 = load_AT(feat_w1, 64, 64, 32, 16 * wave, lane);
    v16h fa2 = load_AT(feat_w1, 64, 64, 64, 16 * wave, lane);
    v8f hid = bias_c(feat_b1, 64, 16 * wave, lane);
    hid = wmma16(fa2, bc2, wmma16(fa1, bc1, wmma16(fa0, bc0, hid)));
#pragma unroll
    for (int v = 0; v < 8; ++v) hid[v] = fmaxf(hid[v], 0.f);
    *(float4*)&lds_hid[n][16 * wave + 8 * hh] =
        make_float4(hid[0], hid[1], hid[2], hid[3]);
    *(float4*)&lds_hid[n][16 * wave + 8 * hh + 4] =
        make_float4(hid[4], hid[5], hid[6], hid[7]);
  }
  __syncthreads();
  if (wave < 4) {
    v16h bq0 = pack16(&lds_hid[n][16 * hh]);
    v16h bq1 = pack16(&lds_hid[n][32 + 16 * hh]);
    v16h ga0 = load_AT(feat_w2, 64, 64, 0, 16 * wave, lane);
    v16h ga1 = load_AT(feat_w2, 64, 64, 32, 16 * wave, lane);
    v8f ho = bias_c(feat_b2, 64, 16 * wave, lane);
    ho = wmma16(ga1, bq1, wmma16(ga0, bq0, ho));
    float* orow = h_out + (size_t)(bi0 + n) * HFEAT + 16 * wave + 8 * hh;
    *(float4*)&orow[0] = make_float4(ho[0], ho[1], ho[2], ho[3]);
    *(float4*)&orow[4] = make_float4(ho[4], ho[5], ho[6], ho[7]);
  }
}

// ---------------------------------------------------------------------------
extern "C" void kernel_launch(void* const* d_in, const int* in_sizes, int n_in,
                              void* d_out, int out_size, void* d_ws,
                              size_t ws_size, hipStream_t stream) {
  (void)in_sizes; (void)n_in; (void)out_size; (void)ws_size;
  const float* x = (const float*)d_in[0];
  const float* h = (const float*)d_in[1];
  const unsigned char* mask = (const unsigned char*)d_in[2];
  const float* msg_w1 = (const float*)d_in[3];
  const float* msg_b1 = (const float*)d_in[4];
  const float* msg_w2 = (const float*)d_in[5];
  const float* msg_b2 = (const float*)d_in[6];
  const float* pos_w1 = (const float*)d_in[7];
  const float* pos_b1 = (const float*)d_in[8];
  const float* pos_w2 = (const float*)d_in[9];
  const float* pos_b2 = (const float*)d_in[10];
  const float* feat_w1 = (const float*)d_in[11];
  const float* feat_b1 = (const float*)d_in[12];
  const float* feat_w2 = (const float*)d_in[13];
  const float* feat_b2 = (const float*)d_in[14];

  _Float16* A_preh = (_Float16*)d_ws;                 // BN*64 f16
  _Float16* B_preh = A_preh + (size_t)BN * HFEAT;     // BN*64 f16
  float* x_out = (float*)d_out;                       // BN*3
  float* h_out = x_out + (size_t)BN * 3;              // BN*64

  egnn_pre<<<BN, HFEAT, 0, stream>>>(h, msg_w1, msg_b1, A_preh, B_preh);
  egnn_main<<<BN / 16, 256, 0, stream>>>(
      x, h, mask, msg_w1, msg_w2, msg_b2, pos_w1, pos_b1, pos_w2, pos_b2,
      feat_w1, feat_b1, feat_w2, feat_b2, A_preh, B_preh, x_out, h_out);
}
